// GQA_70093866270825
// MI455X (gfx1250) — matrix operand order repaired and checked by reading, not compiled, not run on hardware
//
#include <hip/hip_runtime.h>
#include <hip/hip_bf16.h>

// ---------------------------------------------------------------------------
// GQA for MI455X / gfx1250 (wave32, WMMA bf16 16x16x32, f32 accumulate)
//
// D=2048, H=16, G=4, HPG=4, HD=128, B=2, S=2048.
// Pipeline:
//   1) prep: x -> bf16; fuse per-group Wq/Wv into Wqkv -> transposed bf16 Wt[N][K];
//      fused bias bcomb[3D].
//   2) qkv_gemm: (B*S x D) @ (D x 3D) with v_wmma_f32_16x16x32_bf16,
//      64x64 wave tiles, writes q',k,v' bf16 head-major [B,G,HPG,S,HD].
//   3) flash_attn: per (b,head,128-q-rows) streaming softmax attention with
//      post-softmax multiplicative mask; K tile staged to LDS with
//      global_load_async_to_lds_b128 (ASYNCcnt), WMMA for QK^T and PV.
// ---------------------------------------------------------------------------

#define DMODEL 2048
#define SEQ    2048
#define NG     4
#define HPG    4
#define HD     128
#define BATCH  2
#define N3D    (3 * DMODEL)

typedef __attribute__((ext_vector_type(16))) __bf16 v16bf;
typedef __attribute__((ext_vector_type(8)))  float  v8f;

union VFrag {           // one 16x32 bf16 WMMA operand = 32 bytes per lane
    v16bf v;
    uint4 q[2];
};

__device__ __forceinline__ __bf16 f2bf(float f) {
    unsigned u = __float_as_uint(f);
    unsigned r = u + 0x7FFFu + ((u >> 16) & 1u);   // round-to-nearest-even
    unsigned short h = (unsigned short)(r >> 16);
    return __builtin_bit_cast(__bf16, h);
}

__device__ __forceinline__ v8f wmma_bf16(v16bf a, v16bf b, v8f c) {
    // (neg_a, A, neg_b, B, c_mod, C, reuse_a, reuse_b)
    return __builtin_amdgcn_wmma_f32_16x16x32_bf16(false, a, false, b,
                                                   (short)0, c, false, false);
}

// Per-lane async 16B global -> LDS copy (tracked by ASYNCcnt).
__device__ __forceinline__ void async_load_b128(unsigned lds_off,
                                                const void* gptr) {
    asm volatile("global_load_async_to_lds_b128 %0, %1, off"
                 :: "v"(lds_off), "v"(gptr)
                 : "memory");
}

__device__ __forceinline__ void wait_asynccnt0() {
#if __has_builtin(__builtin_amdgcn_s_wait_asynccnt)
    __builtin_amdgcn_s_wait_asynccnt(0);
#else
    asm volatile("s_wait_asynccnt 0x0" ::: "memory");
#endif
}

// ---------------------------------------------------------------------------
// Prep kernels
// ---------------------------------------------------------------------------

__global__ void convert_x_kernel(const float* __restrict__ x,
                                 __bf16* __restrict__ xbf, int n) {
    int i = blockIdx.x * blockDim.x + threadIdx.x;
    if (i < n) xbf[i] = f2bf(x[i]);
}

// Wt[n][c] (bf16, transposed): q/v parts pre-multiplied by per-group Wq/Wv.
__global__ void prep_wt_kernel(const float* __restrict__ Wqkv,
                               const float* __restrict__ Wq,
                               const float* __restrict__ Wv,
                               __bf16* __restrict__ Wt) {
    int idx = blockIdx.x * blockDim.x + threadIdx.x;   // n*2048 + c
    if (idx >= N3D * DMODEL) return;
    int n = idx >> 11;
    int c = idx & 2047;
    int part = n >> 11;          // 0=q, 1=k, 2=v
    int nn   = n & 2047;
    float val;
    if (part == 1) {
        val = Wqkv[(size_t)c * N3D + DMODEL + nn];
    } else {
        int g = nn >> 9, hh = (nn >> 7) & 3, e = nn & 127;
        const float* Wg   = (part == 0 ? Wq : Wv) + g * (HD * HD);
        const float* wrow = Wqkv + (size_t)c * N3D +
                            (part == 0 ? 0 : 2 * DMODEL) + g * 512 + hh * 128;
        float s = 0.f;
        #pragma unroll 8
        for (int d = 0; d < HD; ++d) s = fmaf(wrow[d], Wg[d * HD + e], s);
        val = s;
    }
    Wt[(size_t)n * DMODEL + c] = f2bf(val);
}

__global__ void prep_bias_kernel(const float* __restrict__ bqkv,
                                 const float* __restrict__ Wq,
                                 const float* __restrict__ bq,
                                 const float* __restrict__ Wv,
                                 const float* __restrict__ bv,
                                 float* __restrict__ bcomb) {
    int n = blockIdx.x * blockDim.x + threadIdx.x;
    if (n >= N3D) return;
    int part = n >> 11;
    int nn   = n & 2047;
    float val;
    if (part == 1) {
        val = bqkv[DMODEL + nn];
    } else {
        int g = nn >> 9, hh = (nn >> 7) & 3, e = nn & 127;
        const float* Wg   = (part == 0 ? Wq : Wv) + g * (HD * HD);
        const float* bsm  = (part == 0 ? bq : bv);
        const float* brow = bqkv + (part == 0 ? 0 : 2 * DMODEL) + g * 512 + hh * 128;
        float s = bsm[g * HD + e];
        for (int d = 0; d < HD; ++d) s = fmaf(brow[d], Wg[d * HD + e], s);
        val = s;
    }
    bcomb[n] = val;
}

// ---------------------------------------------------------------------------
// QKV GEMM: C[M=4096][N=6144] = xbf @ Wt^T + bcomb, scatter bf16 head-major.
// Block = 128x256 tile, 8 waves (2 in M, 4 in N), wave tile 64x64:
// 16 WMMAs per 16 b128 loads per K-step.
// ---------------------------------------------------------------------------

__global__ __launch_bounds__(256) void qkv_gemm_kernel(
        const __bf16* __restrict__ xbf, const __bf16* __restrict__ Wt,
        const float* __restrict__ bcomb,
        __bf16* __restrict__ qh, __bf16* __restrict__ kh,
        __bf16* __restrict__ vh) {
    const int tid  = threadIdx.x;
    const int lane = tid & 31, wave = tid >> 5;
    const int half = lane >> 4, ln = lane & 15;
    const int wm = wave & 1, wn = wave >> 1;
    const int m0 = blockIdx.y * 128 + wm * 64;
    const int n0 = blockIdx.x * 256 + wn * 64;

    const v8f vzero = {0.f, 0.f, 0.f, 0.f, 0.f, 0.f, 0.f, 0.f};
    v8f acc[4][4];
    #pragma unroll
    for (int mi = 0; mi < 4; ++mi)
        #pragma unroll
        for (int ni = 0; ni < 4; ++ni) acc[mi][ni] = vzero;

    for (int kk = 0; kk < DMODEL; kk += 32) {
        VFrag a[4];
        #pragma unroll
        for (int mi = 0; mi < 4; ++mi) {
            const __bf16* ar = xbf + (size_t)(m0 + mi * 16 + ln) * DMODEL + kk;
            a[mi].q[0] = *(const uint4*)(ar + half * 8);
            a[mi].q[1] = *(const uint4*)(ar + 16 + half * 8);
            __builtin_prefetch(ar + 128, 0, 3);
        }
        #pragma unroll
        for (int ni = 0; ni < 4; ++ni) {
            const __bf16* br = Wt + (size_t)(n0 + ni * 16 + ln) * DMODEL + kk;
            VFrag bf;
            bf.q[0] = *(const uint4*)(br + half * 8);
            bf.q[1] = *(const uint4*)(br + 16 + half * 8);
            #pragma unroll
            for (int mi = 0; mi < 4; ++mi)
                acc[mi][ni] = wmma_bf16(a[mi].v, bf.v, acc[mi][ni]);
        }
    }

    // Epilogue: bias + scatter to head-major bf16 buffers.
    #pragma unroll
    for (int ni = 0; ni < 4; ++ni) {
        int n = n0 + ni * 16 + ln;
        float bias = bcomb[n];
        int part = n >> 11;
        int nn   = n & 2047;
        int g = nn >> 9, hh = (nn >> 7) & 3, e = nn & 127;
        __bf16* dst = (part == 0) ? qh : ((part == 1) ? kh : vh);
        #pragma unroll
        for (int mi = 0; mi < 4; ++mi) {
            #pragma unroll
            for (int r = 0; r < 8; ++r) {
                int m  = m0 + mi * 16 + half * 8 + r;
                int bb = m >> 11;          // batch
                int s  = m & 2047;         // seq
                dst[((((size_t)bb * NG + g) * HPG + hh) * SEQ + s) * HD + e] =
                    f2bf(acc[mi][ni][r] + bias);
            }
        }
    }
}

// ---------------------------------------------------------------------------
// Flash attention (streaming softmax, post-softmax multiplicative mask).
// Block: one (b, head, 128 query rows); 8 waves x 16 rows each.
// ---------------------------------------------------------------------------

__global__ __launch_bounds__(256) void flash_attn_kernel(
        const __bf16* __restrict__ qh, const __bf16* __restrict__ kh,
        const __bf16* __restrict__ vh, const float* __restrict__ mask,
        float* __restrict__ out) {
    const int tid  = threadIdx.x;
    const int lane = tid & 31, wave = tid >> 5;
    const int half = lane >> 4, ln = lane & 15;
    const int qblk = blockIdx.x;       // 0..15
    const int head = blockIdx.y;       // 0..15 == g*HPG + h
    const int b    = blockIdx.z;       // 0..1
    const size_t headoff = ((size_t)(b * (NG * HPG) + head)) * SEQ * HD;
    const __bf16* Q = qh + headoff;
    const __bf16* K = kh + headoff;
    const __bf16* V = vh + headoff;
    const int qbase = qblk * 128 + wave * 16;

    __shared__ __bf16 Klds[32][HD];       // 8 KB  (keys x headdim, row-major)
    __shared__ __bf16 Vt[HD][32];         // 8 KB  (headdim x keys, transposed)
    __shared__ __bf16 Plds[8][16][32];    // 8 KB  (per-wave probability tile)

    // Q fragments (16 rows x 128 head-dims) resident in registers.
    VFrag qf[4];
    {
        const __bf16* qr = Q + (size_t)(qbase + ln) * HD;
        #pragma unroll
        for (int kc = 0; kc < 4; ++kc) {
            qf[kc].q[0] = *(const uint4*)(qr + kc * 32 + half * 8);
            qf[kc].q[1] = *(const uint4*)(qr + kc * 32 + 16 + half * 8);
        }
    }

    const v8f vzero = {0.f, 0.f, 0.f, 0.f, 0.f, 0.f, 0.f, 0.f};
    v8f acc[8];
    #pragma unroll
    for (int nt = 0; nt < 8; ++nt) acc[nt] = vzero;
    float mrow[8], lrow[8];
    #pragma unroll
    for (int r = 0; r < 8; ++r) { mrow[r] = -1e30f; lrow[r] = 0.f; }

    const float scale = 0.08838834764831845f;   // 1/sqrt(128)

    for (int kb = 0; kb < SEQ; kb += 32) {
        __syncthreads();   // previous-iteration LDS reads done before overwrite
        // K tile: per-lane async global->LDS b128 copies (ASYNCcnt).
        // V tile: manual load + transpose-at-store.
        for (int ch = tid; ch < 512; ch += 256) {
            int row = ch >> 4;            // key 0..31
            int off = (ch & 15) * 8;      // dim chunk
            async_load_b128((unsigned)(size_t)&Klds[row][off],
                            K + (size_t)(kb + row) * HD + off);
            uint4 vd = *(const uint4*)(V + (size_t)(kb + row) * HD + off);
            alignas(16) __bf16 tmp[8];
            *(uint4*)tmp = vd;
            #pragma unroll
            for (int j = 0; j < 8; ++j) Vt[off + j][row] = tmp[j];
        }
        wait_asynccnt0();
        __syncthreads();

        // Scores: S(16x32) = Q(16x128) @ K^T, two 16x16 C tiles.
        v8f sc0 = vzero, sc1 = vzero;
        #pragma unroll
        for (int kc = 0; kc < 4; ++kc) {
            VFrag b0, b1;
            b0.q[0] = *(const uint4*)&Klds[ln][kc * 32 + half * 8];
            b0.q[1] = *(const uint4*)&Klds[ln][kc * 32 + 16 + half * 8];
            b1.q[0] = *(const uint4*)&Klds[16 + ln][kc * 32 + half * 8];
            b1.q[1] = *(const uint4*)&Klds[16 + ln][kc * 32 + 16 + half * 8];
            sc0 = wmma_bf16(qf[kc].v, b0.v, sc0);
            sc1 = wmma_bf16(qf[kc].v, b1.v, sc1);
        }

        // Online softmax per row (rows live across 16 lanes of a half-wave).
        float corr[8];
        #pragma unroll
        for (int r = 0; r < 8; ++r) {
            float s0 = sc0[r] * scale, s1 = sc1[r] * scale;
            float mx = fmaxf(s0, s1);
            #pragma unroll
            for (int o = 8; o >= 1; o >>= 1) mx = fmaxf(mx, __shfl_xor(mx, o, 32));
            float mnew = fmaxf(mrow[r], mx);
            float c = __expf(mrow[r] - mnew);
            mrow[r] = mnew;
            corr[r] = c;
            float p0 = __expf(s0 - mnew), p1 = __expf(s1 - mnew);
            float rs = p0 + p1;                     // UNmasked sum: denominator
            #pragma unroll
            for (int o = 8; o >= 1; o >>= 1) rs += __shfl_xor(rs, o, 32);
            lrow[r] = lrow[r] * c + rs;
            int qrow = qbase + half * 8 + r;
            // post-softmax multiplicative mask applied to numerator only
            float mk0 = mask[(size_t)qrow * SEQ + kb + ln];
            float mk1 = mask[(size_t)qrow * SEQ + kb + 16 + ln];
            Plds[wave][half * 8 + r][ln]      = f2bf(p0 * mk0);
            Plds[wave][half * 8 + r][16 + ln] = f2bf(p1 * mk1);
        }

        // Rescale accumulator by exp(m_old - m_new).
        #pragma unroll
        for (int nt = 0; nt < 8; ++nt)
            #pragma unroll
            for (int r = 0; r < 8; ++r) acc[nt][r] *= corr[r];

        // P in A-fragment layout (round-trip through wave-private LDS).
        VFrag pf;
        pf.q[0] = *(const uint4*)&Plds[wave][ln][half * 8];
        pf.q[1] = *(const uint4*)&Plds[wave][ln][16 + half * 8];

        // acc(16x128) += P(16x32) @ V(32x128)
        #pragma unroll
        for (int nt = 0; nt < 8; ++nt) {
            VFrag bf;
            bf.q[0] = *(const uint4*)&Vt[nt * 16 + ln][half * 8];
            bf.q[1] = *(const uint4*)&Vt[nt * 16 + ln][16 + half * 8];
            acc[nt] = wmma_bf16(pf.v, bf.v, acc[nt]);
        }
    }

    // Epilogue: out = acc / l, fp32 to [B,S,D] with D index = head*128 + e.
    #pragma unroll
    for (int r = 0; r < 8; ++r) {
        float inv = 1.0f / lrow[r];
        int qrow = qbase + half * 8 + r;
        float* orow = out + ((size_t)b * SEQ + qrow) * DMODEL + (size_t)head * HD;
        #pragma unroll
        for (int nt = 0; nt < 8; ++nt) orow[nt * 16 + ln] = acc[nt][r] * inv;
    }
}

// ---------------------------------------------------------------------------
// Launch
// ---------------------------------------------------------------------------

extern "C" void kernel_launch(void* const* d_in, const int* in_sizes, int n_in,
                              void* d_out, int out_size, void* d_ws, size_t ws_size,
                              hipStream_t stream) {
    (void)in_sizes; (void)n_in; (void)out_size; (void)ws_size;
    const float* x     = (const float*)d_in[0];
    const float* amask = (const float*)d_in[1];
    const float* Wqkv  = (const float*)d_in[2];
    const float* bqkv  = (const float*)d_in[3];
    const float* Wq    = (const float*)d_in[4];
    const float* bq    = (const float*)d_in[5];
    const float* Wv    = (const float*)d_in[6];
    const float* bv    = (const float*)d_in[7];
    float* out = (float*)d_out;
    char*  ws  = (char*)d_ws;

    // Workspace layout (bytes)
    __bf16* xbf   = (__bf16*)(ws);                    // 16,777,216
    __bf16* Wt    = (__bf16*)(ws + 16777216ull);      // 25,165,824
    float*  bcomb = (float*) (ws + 41943040ull);      //     24,576
    __bf16* qh    = (__bf16*)(ws + 41967616ull);      // 16,777,216
    __bf16* kh    = (__bf16*)(ws + 58744832ull);      // 16,777,216
    __bf16* vh    = (__bf16*)(ws + 75522048ull);      // 16,777,216
                                                      // total ~92.3 MB

    const int NX = BATCH * SEQ * DMODEL;              // 8,388,608
    convert_x_kernel<<<(NX + 255) / 256, 256, 0, stream>>>(x, xbf, NX);
    prep_wt_kernel<<<(N3D * DMODEL + 255) / 256, 256, 0, stream>>>(Wqkv, Wq, Wv, Wt);
    prep_bias_kernel<<<(N3D + 255) / 256, 256, 0, stream>>>(bqkv, Wq, bq, Wv, bv, bcomb);

    qkv_gemm_kernel<<<dim3(N3D / 256, (BATCH * SEQ) / 128), 256, 0, stream>>>(
        xbf, Wt, bcomb, qh, kh, vh);

    flash_attn_kernel<<<dim3(SEQ / 128, NG * HPG, BATCH), 256, 0, stream>>>(
        qh, kh, vh, amask, out);
}
